// DEQLayer_15496242004634
// MI455X (gfx1250) — matrix operand order repaired and checked by reading, not compiled, not run on hardware
//
#include <hip/hip_runtime.h>
#include <hip/hip_bf16.h>

typedef __bf16 bf16_t;
typedef __attribute__((ext_vector_type(4)))  float   f32x4;
typedef __attribute__((ext_vector_type(4)))  bf16_t  v4bf;
typedef __attribute__((ext_vector_type(8)))  float   v8f;
typedef __attribute__((ext_vector_type(8)))  bf16_t  v8bf;
typedef __attribute__((ext_vector_type(16))) bf16_t  v16bf;

#define BROWS     64            // rows per workgroup
#define NTHREADS  256           // 8 x wave32; wave pairs split N
#define HD        256
#define INDIM     48
#define KPAD_X    64
#define ODIM      32
#define ASTRIDE   264           // activation LDS row stride (elems), breaks bank conflicts
#define WSTRIDE   264           // weight LDS row stride for K=256
#define XSTRIDE   72            // row stride for K=64 buffers

// workspace layout (bf16 elements): weights pre-converted + transposed (N-major)
#define WS_OFF_WI 0                       // [256][64]  (K padded 48->64)
#define WS_OFF_W1 (WS_OFF_WI + 256 * 64)  // [256][256]
#define WS_OFF_W2 (WS_OFF_W1 + 256 * 256) // [256][256]
#define WS_OFF_WO (WS_OFF_W2 + 256 * 256) // [32][256]

// LDS layout (bytes)
#define OFF_W     0
#define SZ_W      (256 * WSTRIDE * 2)      // 135168
#define OFF_XINP  (OFF_W + SZ_W)
#define SZ_ACT    (BROWS * ASTRIDE * 2)    // 33792
#define OFF_Z1    (OFF_XINP + SZ_ACT)
#define OFF_ACT   (OFF_Z1 + SZ_ACT)
#define OFF_XA    (OFF_ACT + SZ_ACT)
#define SZ_XA     (BROWS * XSTRIDE * 2)    // 9216
#define OFF_RED   (OFF_XA + SZ_XA)         // cross-wave LN reduction buffer
#define SZ_RED    (BROWS * 4 * 4)          // 64 rows x {s,q} x 2 halves (f32)
#define SMEM_BYTES (OFF_RED + SZ_RED)      // 246784 (< 320KB WGP LDS)

// ---- async global->LDS copy (gfx1250 GLOBAL_LOAD_ASYNC_TO_LDS_B128) -------
__device__ __forceinline__ void async_copy_w(const bf16_t* __restrict__ g,
                                             int gStrideElems, bf16_t* sDst,
                                             int sStrideElems, int nRows,
                                             int chunkShift /* log2(chunks/row) */,
                                             int tid) {
  const int total = nRows << chunkShift;
  const int cmask = (1 << chunkShift) - 1;
  for (int c = tid; c < total; c += NTHREADS) {
    const int r  = c >> chunkShift;
    const int cc = c & cmask;
    const bf16_t* gp = g + r * gStrideElems + (cc << 3);
    unsigned lds = (unsigned)(size_t)(void*)(sDst + r * sStrideElems + (cc << 3));
    asm volatile("global_load_async_to_lds_b128 %0, %1, off"
                 :: "v"(lds), "v"(gp) : "memory");
  }
}
__device__ __forceinline__ void wait_async0() {
  asm volatile("s_wait_asynccnt 0x0" ::: "memory");
}

// ---- WMMA operand fragment loaders (ISA 7.12.2 layouts) -------------------
// A (16x32 bf16): lanes 0-15 row M=lane, K = kb+0..7 then kb+16..23
//                 lanes 16-31 row M=lane-16, K = kb+8..15 then kb+24..31
__device__ __forceinline__ v16bf load_a_frag(const bf16_t* S, int stride,
                                             int rowBase, int kb, int lane) {
  const int row  = rowBase + (lane & 15);
  const int koff = (lane & 16) ? 8 : 0;
  const bf16_t* p = S + row * stride + kb + koff;
  union { v16bf v; v8bf h[2]; } u;
  u.h[0] = *(const v8bf*)(p);
  u.h[1] = *(const v8bf*)(p + 16);
  return u.v;
}
// B (32x16 bf16), weights stored N-major [n][k]:
// lanes 0-15 col N=lane, K = kb+0..7 then kb+16..23 ; lanes 16-31 col N=lane-16, K = kb+8..15 then kb+24..31
__device__ __forceinline__ v16bf load_b_frag(const bf16_t* Wt, int stride,
                                             int nBase, int kb, int lane) {
  const int n    = nBase + (lane & 15);
  const int koff = (lane & 16) ? 8 : 0;
  const bf16_t* p = Wt + n * stride + kb + koff;
  union { v16bf v; v8bf h[2]; } u;
  u.h[0] = *(const v8bf*)(p);
  u.h[1] = *(const v8bf*)(p + 16);
  return u.v;
}

// ---- 16-row x 128-col GEMM tile per wave (wave pair covers 256 cols) ------
template <int KTOT>
__device__ __forceinline__ void gemm16x128(v8f acc[8], const bf16_t* A, int strideA,
                                           int rowBase, const bf16_t* Wt, int strideW,
                                           int tBase, int lane) {
#pragma unroll
  for (int kb = 0; kb < KTOT; kb += 32) {
    v16bf a = load_a_frag(A, strideA, rowBase, kb, lane);
#pragma unroll
    for (int t = 0; t < 8; ++t) {
      v16bf b = load_b_frag(Wt, strideW, (tBase + t) * 16, kb, lane);
      acc[t] = __builtin_amdgcn_wmma_f32_16x16x32_bf16(
          false, a, false, b, (short)0, acc[t], false, false);
    }
  }
}

// ---- bias (+opt residual) (+opt relu) then LayerNorm over the wave pair ---
// C/D layout: lane L, VGPR v -> row M = v + 8*(L>>4), col N = tile*16 + (L&15)
template <bool RELU, bool ADD>
__device__ __forceinline__ void epilogue_ln(v8f acc[8], int lane, int waveRow,
                                            int tBase,
                                            const float* __restrict__ bias,
                                            const float* __restrict__ gamma,
                                            const float* __restrict__ beta,
                                            const bf16_t* __restrict__ addBuf,
                                            bf16_t* __restrict__ dst,
                                            float* __restrict__ sRed) {
  const int half  = lane >> 4;
  const int nlo   = lane & 15;
  const int nHalf = tBase >> 3;   // 0 or 1
#pragma unroll
  for (int t = 0; t < 8; ++t) {
    const int n = (tBase + t) * 16 + nlo;
    const float bn = bias[n];
#pragma unroll
    for (int v = 0; v < 8; ++v) {
      float val = acc[t][v] + bn;
      if (ADD) val += (float)addBuf[(waveRow + v + 8 * half) * ASTRIDE + n];
      if (RELU) val = fmaxf(val, 0.0f);
      acc[t][v] = val;
    }
  }
  // per-row partial sums over this wave's 128 columns
#pragma unroll
  for (int v = 0; v < 8; ++v) {
    float s = 0.0f, q = 0.0f;
#pragma unroll
    for (int t = 0; t < 8; ++t) { float a = acc[t][v]; s += a; q += a * a; }
    s += __shfl_xor(s, 1);  q += __shfl_xor(q, 1);
    s += __shfl_xor(s, 2);  q += __shfl_xor(q, 2);
    s += __shfl_xor(s, 4);  q += __shfl_xor(q, 4);
    s += __shfl_xor(s, 8);  q += __shfl_xor(q, 8);
    if (nlo == 0) {
      const int row = waveRow + v + 8 * half;
      sRed[row * 4 + nHalf * 2 + 0] = s;
      sRed[row * 4 + nHalf * 2 + 1] = q;
    }
  }
  __syncthreads();   // exchange partials between the wave pair
  float mean[8], rstd[8];
#pragma unroll
  for (int v = 0; v < 8; ++v) {
    const int row = waveRow + v + 8 * half;
    float s = sRed[row * 4 + 0] + sRed[row * 4 + 2];
    float q = sRed[row * 4 + 1] + sRed[row * 4 + 3];
    float m = s * (1.0f / 256.0f);
    mean[v] = m;
    rstd[v] = rsqrtf(q * (1.0f / 256.0f) - m * m + 1e-5f);
  }
#pragma unroll
  for (int t = 0; t < 8; ++t) {
    const int n = (tBase + t) * 16 + nlo;
    const float g = gamma[n], bb = beta[n];
#pragma unroll
    for (int v = 0; v < 8; ++v) {
      float y = (acc[t][v] - mean[v]) * rstd[v] * g + bb;
      dst[(waveRow + v + 8 * half) * ASTRIDE + n] = (bf16_t)y;
    }
  }
}

// ---- prep: convert + transpose weights to bf16 (N-major) in workspace -----
__global__ __launch_bounds__(256) void prep_weights(
    const float* __restrict__ W_inp, const float* __restrict__ W1,
    const float* __restrict__ W2, const float* __restrict__ W_out,
    bf16_t* __restrict__ ws) {
  bf16_t* Wit = ws + WS_OFF_WI;
  bf16_t* W1t = ws + WS_OFF_W1;
  bf16_t* W2t = ws + WS_OFF_W2;
  bf16_t* Wot = ws + WS_OFF_WO;
  const int stride = gridDim.x * blockDim.x;
  const int t0 = blockIdx.x * blockDim.x + threadIdx.x;
  for (int i = t0; i < 256 * KPAD_X; i += stride) {
    int n = i >> 6, k = i & 63;
    Wit[n * KPAD_X + k] = (k < INDIM) ? (bf16_t)W_inp[k * HD + n] : (bf16_t)0.0f;
  }
  for (int i = t0; i < HD * HD; i += stride) {
    int k = i >> 8, n = i & 255;   // coalesced read W[i], transposed write
    W1t[n * HD + k] = (bf16_t)W1[i];
    W2t[n * HD + k] = (bf16_t)W2[i];
  }
  for (int i = t0; i < HD * ODIM; i += stride) {
    int k = i >> 5, n = i & 31;
    Wot[n * HD + k] = (bf16_t)W_out[i];
  }
}

// ---------------------------------------------------------------------------
__global__ __launch_bounds__(NTHREADS) void deq_fused(
    const float* __restrict__ x, const float* __restrict__ z,
    const bf16_t* __restrict__ ws,
    const float* __restrict__ b_inp, const float* __restrict__ g_ln_inp,
    const float* __restrict__ b_ln_inp, const float* __restrict__ b1,
    const float* __restrict__ g_ln1, const float* __restrict__ b_ln1,
    const float* __restrict__ b2, const float* __restrict__ g_ln2,
    const float* __restrict__ b_ln2, const float* __restrict__ g_ln3,
    const float* __restrict__ b_ln3, const float* __restrict__ b_out,
    float* __restrict__ dx, float* __restrict__ zout, int nrows) {
  extern __shared__ char smem[];
  bf16_t* sW    = (bf16_t*)(smem + OFF_W);     // weight buffer (reused per stage)
  bf16_t* sXinp = (bf16_t*)(smem + OFF_XINP);  // xinp bf16, later z_out bf16
  bf16_t* sZ1   = (bf16_t*)(smem + OFF_Z1);    // z1 bf16
  bf16_t* sAct  = (bf16_t*)(smem + OFF_ACT);   // z bf16, later inner bf16
  bf16_t* sXa   = (bf16_t*)(smem + OFF_XA);    // x bf16 (K padded to 64)
  float*  sRed  = (float*)(smem + OFF_RED);    // LN partial sums

  const int tid      = threadIdx.x;
  const int lane     = tid & 31;
  const int wave     = tid >> 5;           // 0..7
  const int waveRow  = (wave >> 1) * 16;   // 4 row-tiles of 16
  const int nHalf    = wave & 1;           // wave pair splits N
  const int tBase    = nHalf * 8;          // 8 N-tiles per wave
  const int blockRow = blockIdx.x * BROWS;

  // ---- kick off W_inp^T DMA; stage x and z (f32 -> bf16) meanwhile ----
  async_copy_w(ws + WS_OFF_WI, KPAD_X, sW, XSTRIDE, 256, /*chunks/row=8*/ 3, tid);
  for (int idx = tid; idx < BROWS * 16; idx += NTHREADS) {   // x: 12 f32x4 + pad
    int r = idx >> 4, c4 = idx & 15;
    int gr = min(blockRow + r, nrows - 1);
    f32x4 v = {0.0f, 0.0f, 0.0f, 0.0f};
    if (c4 < 12) v = *(const f32x4*)(x + (size_t)gr * INDIM + c4 * 4);
    v4bf o;
    o.x = (bf16_t)v.x; o.y = (bf16_t)v.y; o.z = (bf16_t)v.z; o.w = (bf16_t)v.w;
    *(v4bf*)(sXa + r * XSTRIDE + c4 * 4) = o;
  }
  for (int idx = tid; idx < BROWS * (HD / 4); idx += NTHREADS) {
    int r = idx >> 6, n4 = idx & 63;
    int gr = min(blockRow + r, nrows - 1);
    f32x4 v = *(const f32x4*)(z + (size_t)gr * HD + n4 * 4);
    v4bf o;
    o.x = (bf16_t)v.x; o.y = (bf16_t)v.y; o.z = (bf16_t)v.z; o.w = (bf16_t)v.w;
    *(v4bf*)(sAct + r * ASTRIDE + n4 * 4) = o;
  }
  wait_async0();
  __syncthreads();

  // ---- stage A: xinp = LN(x @ W_inp + b_inp) ----
  {
    v8f acc[8];
#pragma unroll
    for (int t = 0; t < 8; ++t)
#pragma unroll
      for (int v = 0; v < 8; ++v) acc[t][v] = 0.0f;
    gemm16x128<KPAD_X>(acc, sXa, XSTRIDE, waveRow, sW, XSTRIDE, tBase, lane);
    __syncthreads();  // all waves done reading sW (W_inp^T)
    async_copy_w(ws + WS_OFF_W1, HD, sW, WSTRIDE, 256, /*chunks/row=32*/ 5, tid);
    epilogue_ln<false, false>(acc, lane, waveRow, tBase, b_inp, g_ln_inp,
                              b_ln_inp, nullptr, sXinp, sRed);
  }
  wait_async0();
  __syncthreads();

  // ---- stage B: z1 = LN(relu(z @ W1 + b1)) ----
  {
    v8f acc[8];
#pragma unroll
    for (int t = 0; t < 8; ++t)
#pragma unroll
      for (int v = 0; v < 8; ++v) acc[t][v] = 0.0f;
    gemm16x128<HD>(acc, sAct, ASTRIDE, waveRow, sW, WSTRIDE, tBase, lane);
    __syncthreads();  // all waves done reading sW (W1^T)
    async_copy_w(ws + WS_OFF_W2, HD, sW, WSTRIDE, 256, 5, tid);
    epilogue_ln<true, false>(acc, lane, waveRow, tBase, b1, g_ln1, b_ln1,
                             nullptr, sZ1, sRed);
  }
  wait_async0();
  __syncthreads();

  // ---- stage C: inner = LN(xinp + z1 @ W2 + b2) -> sAct ----
  {
    v8f acc[8];
#pragma unroll
    for (int t = 0; t < 8; ++t)
#pragma unroll
      for (int v = 0; v < 8; ++v) acc[t][v] = 0.0f;
    gemm16x128<HD>(acc, sZ1, ASTRIDE, waveRow, sW, WSTRIDE, tBase, lane);
    __syncthreads();  // all waves done reading sW (W2^T)
    async_copy_w(ws + WS_OFF_WO, HD, sW, WSTRIDE, 32, 5, tid);
    epilogue_ln<false, true>(acc, lane, waveRow, tBase, b2, g_ln2, b_ln2,
                             sXinp, sAct, sRed);
  }
  wait_async0();
  __syncthreads();

  // ---- stage D: z_out = LN(relu(z1 + inner)); coalesced f32x4 stores ----
  {
    const int col = lane * 8;     // 8 cols per lane, wave covers a full row
    const f32x4 g0 = *(const f32x4*)(g_ln3 + col);
    const f32x4 g1 = *(const f32x4*)(g_ln3 + col + 4);
    const f32x4 be0 = *(const f32x4*)(b_ln3 + col);
    const f32x4 be1 = *(const f32x4*)(b_ln3 + col + 4);
#pragma unroll
    for (int r0 = 0; r0 < 8; ++r0) {
      const int row = wave * 8 + r0;
      v8bf z1v = *(const v8bf*)(sZ1 + row * ASTRIDE + col);
      v8bf inv = *(const v8bf*)(sAct + row * ASTRIDE + col);
      float val[8];
      float s = 0.0f, q = 0.0f;
#pragma unroll
      for (int i = 0; i < 8; ++i) {
        float v = (float)z1v[i] + (float)inv[i];
        v = fmaxf(v, 0.0f);
        val[i] = v; s += v; q += v * v;
      }
      s += __shfl_xor(s, 1);  q += __shfl_xor(q, 1);
      s += __shfl_xor(s, 2);  q += __shfl_xor(q, 2);
      s += __shfl_xor(s, 4);  q += __shfl_xor(q, 4);
      s += __shfl_xor(s, 8);  q += __shfl_xor(q, 8);
      s += __shfl_xor(s, 16); q += __shfl_xor(q, 16);
      float mean = s * (1.0f / 256.0f);
      float rs   = rsqrtf(q * (1.0f / 256.0f) - mean * mean + 1e-5f);
      f32x4 o0, o1;
      v8bf ob;
#pragma unroll
      for (int i = 0; i < 8; ++i) {
        float g = (i < 4) ? g0[i] : g1[i - 4];
        float bb = (i < 4) ? be0[i] : be1[i - 4];
        float y = (val[i] - mean) * rs * g + bb;
        if (i < 4) o0[i] = y; else o1[i - 4] = y;
        ob[i] = (bf16_t)y;
      }
      const int gr = blockRow + row;
      if (gr < nrows) {
        *(f32x4*)(zout + (size_t)gr * HD + col)     = o0;
        *(f32x4*)(zout + (size_t)gr * HD + col + 4) = o1;
      }
      *(v8bf*)(sXinp + row * ASTRIDE + col) = ob;   // A operand for fc_out
    }
  }
  __syncthreads();

  // ---- stage E: dx = z_out @ W_out + b_out (one 16x16 tile per wave) ----
  {
    v8f acc2;
#pragma unroll
    for (int v = 0; v < 8; ++v) acc2[v] = 0.0f;
#pragma unroll
    for (int kb = 0; kb < HD; kb += 32) {
      v16bf a = load_a_frag(sXinp, ASTRIDE, waveRow, kb, lane);
      v16bf b = load_b_frag(sW, WSTRIDE, nHalf * 16, kb, lane);
      acc2 = __builtin_amdgcn_wmma_f32_16x16x32_bf16(
          false, a, false, b, (short)0, acc2, false, false);
    }
    const int half = lane >> 4;
    const int nlo  = lane & 15;
    const int n    = nHalf * 16 + nlo;
    const float bo = b_out[n];
#pragma unroll
    for (int v = 0; v < 8; ++v) {
      const int gr = blockRow + waveRow + v + 8 * half;
      if (gr < nrows) dx[(size_t)gr * ODIM + n] = acc2[v] + bo;
    }
  }
}

extern "C" void kernel_launch(void* const* d_in, const int* in_sizes, int n_in,
                              void* d_out, int out_size, void* d_ws, size_t ws_size,
                              hipStream_t stream) {
  const float* x        = (const float*)d_in[0];
  const float* z        = (const float*)d_in[1];
  const float* W_inp    = (const float*)d_in[2];
  const float* b_inp    = (const float*)d_in[3];
  const float* g_ln_inp = (const float*)d_in[4];
  const float* b_ln_inp = (const float*)d_in[5];
  const float* W1       = (const float*)d_in[6];
  const float* b1       = (const float*)d_in[7];
  const float* g_ln1    = (const float*)d_in[8];
  const float* b_ln1    = (const float*)d_in[9];
  const float* W2       = (const float*)d_in[10];
  const float* b2       = (const float*)d_in[11];
  const float* g_ln2    = (const float*)d_in[12];
  const float* b_ln2    = (const float*)d_in[13];
  const float* g_ln3    = (const float*)d_in[14];
  const float* b_ln3    = (const float*)d_in[15];
  const float* W_out    = (const float*)d_in[16];
  const float* b_out    = (const float*)d_in[17];

  const int B = in_sizes[0] / INDIM;
  float* dx   = (float*)d_out;                 // (B, 32) flat, first output
  float* zo   = dx + (size_t)B * ODIM;         // (B, 256), second output
  bf16_t* ws  = (bf16_t*)d_ws;                 // ~304KB of bf16 weights

  prep_weights<<<64, 256, 0, stream>>>(W_inp, W1, W2, W_out, ws);

  const int nblocks = (B + BROWS - 1) / BROWS;
  deq_fused<<<nblocks, NTHREADS, SMEM_BYTES, stream>>>(
      x, z, ws, b_inp, g_ln_inp, b_ln_inp, b1, g_ln1, b_ln1,
      b2, g_ln2, b_ln2, g_ln3, b_ln3, b_out, dx, zo, B);
}
